// GraphProp_10565619548251
// MI455X (gfx1250) — matrix operand order repaired and checked by reading, not compiled, not run on hardware
//
#include <hip/hip_runtime.h>

#define N_NODES 50000
#define N_EDGES 800000
#define HDIM    128

typedef __attribute__((ext_vector_type(16))) __bf16 v16bf;
typedef __attribute__((ext_vector_type(8)))  float  v8f;

union Frag16 {
  v16bf v;
  uint4 q[2];
  unsigned short h[16];
};

static __device__ __forceinline__ unsigned short f32_to_bf16_rne(float f) {
  unsigned int u = __float_as_uint(f);
  u += 0x7FFFu + ((u >> 16) & 1u);
  return (unsigned short)(u >> 16);
}

static __device__ __forceinline__ float sigmoid_f(float x) {
  return 1.0f / (1.0f + __expf(-x));
}

static __device__ __forceinline__ v8f wmma_bf16(v16bf a, v16bf b, v8f c) {
  return __builtin_amdgcn_wmma_f32_16x16x32_bf16(false, a, false, b, (short)0, c, false, false);
}

static __device__ __forceinline__ void pack8(Frag16& F, int base, float4 a, float4 b) {
  F.h[base + 0] = f32_to_bf16_rne(a.x); F.h[base + 1] = f32_to_bf16_rne(a.y);
  F.h[base + 2] = f32_to_bf16_rne(a.z); F.h[base + 3] = f32_to_bf16_rne(a.w);
  F.h[base + 4] = f32_to_bf16_rne(b.x); F.h[base + 5] = f32_to_bf16_rne(b.y);
  F.h[base + 6] = f32_to_bf16_rne(b.z); F.h[base + 7] = f32_to_bf16_rne(b.w);
}

static __device__ __forceinline__ void load_frag_g(const uint4* base, int fi, int lane, Frag16& F) {
  const uint4* p = base + (long long)(fi * 32 + lane) * 2;
  F.q[0] = p[0];
  F.q[1] = p[1];
}

// ---------------- utility kernels ----------------

__global__ void k_cvt_bf16(const float* __restrict__ in, unsigned short* __restrict__ out, int n) {
  int i = blockIdx.x * blockDim.x + threadIdx.x;
  if (i < n) out[i] = f32_to_bf16_rne(in[i]);
}

__global__ void k_zero4(float4* __restrict__ p, int n4) {
  int i = blockIdx.x * blockDim.x + threadIdx.x;
  if (i < n4) p[i] = make_float4(0.f, 0.f, 0.f, 0.f);
}

// Pack W[o][k] (row-major, row pitch `pitch`, fp32) into bf16 WMMA B-fragments.
// out[(((nt*ksteps)+ks)*32 + lane)*16 + e] = bf16(W[nt*16 + (lane&15)][ks*32 + (lane>>4)*8 + (e&7) + (e>>3)*16])
__global__ void k_pack_bfrag(const float* __restrict__ W, int pitch, int ksteps, int total,
                             unsigned short* __restrict__ out) {
  int idx = blockIdx.x * blockDim.x + threadIdx.x;
  if (idx >= total) return;
  int e    = idx & 15;
  int lane = (idx >> 4) & 31;
  int ks   = (idx >> 9) % ksteps;
  int nt   = idx / (ksteps << 9);
  int o = nt * 16 + (lane & 15);
  int k = ks * 32 + ((lane >> 4) << 3) + (e & 7) + ((e >> 3) << 4);
  out[idx] = f32_to_bf16_rne(W[(long long)o * pitch + k]);
}

// Pack W_msg (256x257) into 16nt x 9ks fragments; the 9th k-step carries
// k=256 -> W_msg[:,256] (he column weight) and k=257 -> b_msg (bias), zeros elsewhere.
__global__ void k_pack_wmsg(const float* __restrict__ W,     // [256][257]
                            const float* __restrict__ bmsg,  // [256]
                            unsigned short* __restrict__ out) {
  int idx = blockIdx.x * blockDim.x + threadIdx.x;
  if (idx >= 16 * 9 * 512) return;
  int e    = idx & 15;
  int lane = (idx >> 4) & 31;
  int ks   = (idx >> 9) % 9;
  int nt   = idx / (9 << 9);
  int o    = nt * 16 + (lane & 15);
  int half = lane >> 4;
  unsigned short v;
  if (ks < 8) {
    int k = ks * 32 + (half << 3) + (e & 7) + ((e >> 3) << 4);
    v = f32_to_bf16_rne(W[(long long)o * 257 + k]);
  } else if (half == 0 && e == 0) {
    v = f32_to_bf16_rne(W[(long long)o * 257 + 256]);   // he weight
  } else if (half == 0 && e == 1) {
    v = f32_to_bf16_rne(bmsg[o]);                       // bias (A supplies 1.0)
  } else {
    v = 0;
  }
  out[idx] = v;
}

// ---------------- edge message GEMM + segment-sum ----------------
// One wave handles 16 edges. K = 288: 8 k-steps of [hv[dst] || hv[src]] plus a 9th
// k-step carrying [he, 1, 0...], so bias + he-column are folded into the WMMA chain.
// W fragments staged in LDS (144 KB, shared by 8 waves); B feed software-pipelined.
// Epilogue is pure f32 atomic segment-sum into a_acc[dst].

__global__ void __launch_bounds__(256)
k_edge(const unsigned short* __restrict__ hv_bf,   // [N][128] bf16
       const float* __restrict__ he,               // [E]
       const int*   __restrict__ src,
       const int*   __restrict__ dst,
       const unsigned short* __restrict__ wfrag,   // packed 16nt x 9ks frags
       float* __restrict__ a_acc)                  // [N][256] f32
{
  __shared__ uint4 ldsW[9216];                     // 144 KB: 144 frags * 32 lanes * 32B

  const int tid = threadIdx.x;
  {
    const uint4* wf = (const uint4*)wfrag;
#pragma unroll
    for (int i = 0; i < 36; ++i)
      ldsW[tid + i * 256] = wf[tid + i * 256];
  }
  __syncthreads();

  const int lane  = tid & 31;
  const int wv    = tid >> 5;
  const int tile  = blockIdx.x * 8 + wv;           // 0 .. 49999
  const int ebase = tile * 16;
  const int half  = lane >> 4;
  const int mrow  = lane & 15;

  const int ndst = dst[ebase + mrow];
  const int nsrc = src[ebase + mrow];

  // A fragments: k-steps 0..3 from hv[dst], 4..7 from hv[src], 8 = [he, 1, 0...]
  Frag16 A[9];
#pragma unroll
  for (int ks = 0; ks < 8; ++ks) {
    int node = (ks < 4) ? ndst : nsrc;
    const char* rowp = (const char*)hv_bf + (long long)node * 256 + (ks & 3) * 64 + half * 16;
    A[ks].q[0] = *(const uint4*)(rowp);
    A[ks].q[1] = *(const uint4*)(rowp + 32);
  }
  {
    unsigned short heb = f32_to_bf16_rne(he[ebase + mrow]);
    A[8].q[0] = make_uint4(0u, 0u, 0u, 0u);
    A[8].q[1] = make_uint4(0u, 0u, 0u, 0u);
    A[8].h[0] = (half == 0) ? heb : (unsigned short)0;          // k = 256
    A[8].h[1] = (half == 0) ? (unsigned short)0x3F80 : (unsigned short)0;  // k = 257 (1.0)
  }

  // Epilogue row data: this lane owns D rows m = half*8 + r
  const int mr = half * 8;
  int drow[8];
#pragma unroll
  for (int r = 0; r < 8; ++r)
    drow[r] = dst[ebase + mr + r];

  const int ncol = lane & 15;

#pragma unroll 1
  for (int nt = 0; nt < 16; ++nt) {
    v8f acc = {0.f, 0.f, 0.f, 0.f, 0.f, 0.f, 0.f, 0.f};
    const int fbase = nt * 9;

    Frag16 B[3];
    {
      int i0 = ((fbase + 0) * 32 + lane) * 2;
      B[0].q[0] = ldsW[i0]; B[0].q[1] = ldsW[i0 + 1];
      int i1 = ((fbase + 1) * 32 + lane) * 2;
      B[1].q[0] = ldsW[i1]; B[1].q[1] = ldsW[i1 + 1];
    }
#pragma unroll
    for (int ks = 0; ks < 9; ++ks) {
      if (ks < 7) {                                // depth-2 prefetch
        int ip = ((fbase + ks + 2) * 32 + lane) * 2;
        int s  = (ks + 2) % 3;
        B[s].q[0] = ldsW[ip]; B[s].q[1] = ldsW[ip + 1];
      }
      acc = wmma_bf16(A[ks].v, B[ks % 3].v, acc);
    }

    const int o = nt * 16 + ncol;
#pragma unroll
    for (int r = 0; r < 8; ++r)
      unsafeAtomicAdd(&a_acc[(long long)drow[r] * 256 + o], acc[r]);   // segment-sum
  }
}

// ---------------- GRU cell ----------------
// One wave handles 16 nodes. gi = a@Wih^T (K=256), gh = h@Whh^T (K=128), then gates.
// B-frag trios are double-buffered across a fused 12-step (8 ih + 4 hh) pipeline.

__global__ void __launch_bounds__(160)
k_gru(const float* __restrict__ a_acc,             // [N][256] f32
      const unsigned short* __restrict__ hv_bf,    // [N][128] bf16 (current h)
      const float* __restrict__ hv_cur,            // [N][128] f32  (current h)
      const unsigned short* __restrict__ wih_frag, // 24nt x 8ks
      const unsigned short* __restrict__ whh_frag, // 24nt x 4ks
      const float* __restrict__ b_ih,              // [384]
      const float* __restrict__ b_hh,              // [384]
      float* __restrict__ hv_out,                  // [N][128] f32
      unsigned short* hv_bf_out)                   // [N][128] bf16 (may alias hv_bf)
{
  const int lane  = threadIdx.x & 31;
  const int wv    = threadIdx.x >> 5;
  const int tile  = blockIdx.x * 5 + wv;           // 0 .. 3124
  const int nbase = tile * 16;
  const int half  = lane >> 4;
  const int mrow  = lane & 15;

  // A fragments for gi: aggregated messages, f32 -> bf16 on the fly
  Frag16 Aa[8];
  {
    const float* arow = a_acc + (long long)(nbase + mrow) * 256;
#pragma unroll
    for (int ks = 0; ks < 8; ++ks) {
      int k0 = ks * 32 + half * 8;
      float4 f0 = *(const float4*)(arow + k0);
      float4 f1 = *(const float4*)(arow + k0 + 4);
      float4 f2 = *(const float4*)(arow + k0 + 16);
      float4 f3 = *(const float4*)(arow + k0 + 20);
      pack8(Aa[ks], 0, f0, f1);
      pack8(Aa[ks], 8, f2, f3);
    }
  }
  // A fragments for gh: current h (bf16)
  Frag16 Ah[4];
#pragma unroll
  for (int ks = 0; ks < 4; ++ks) {
    const char* rowp = (const char*)hv_bf + (long long)(nbase + mrow) * 256 + ks * 64 + half * 16;
    Ah[ks].q[0] = *(const uint4*)(rowp);
    Ah[ks].q[1] = *(const uint4*)(rowp + 32);
  }

  const int ncol = lane & 15;
  const int mr   = half * 8;
  const uint4* wih = (const uint4*)wih_frag;
  const uint4* whh = (const uint4*)whh_frag;

#pragma unroll 1
  for (int g = 0; g < 8; ++g) {                    // gate tile over H=128
    v8f ir = {0.f,0.f,0.f,0.f,0.f,0.f,0.f,0.f};
    v8f iz = ir, in_ = ir, hr = ir, hz = ir, hn = ir;

    Frag16 T[2][3];
    load_frag_g(wih, (g)      * 8 + 0, lane, T[0][0]);
    load_frag_g(wih, (g + 8)  * 8 + 0, lane, T[0][1]);
    load_frag_g(wih, (g + 16) * 8 + 0, lane, T[0][2]);

#pragma unroll
    for (int s = 0; s < 12; ++s) {                 // 8 ih steps + 4 hh steps, fused pipeline
      const int cb = s & 1;
      const int nb = cb ^ 1;
      if (s < 7) {
        load_frag_g(wih, (g)      * 8 + s + 1, lane, T[nb][0]);
        load_frag_g(wih, (g + 8)  * 8 + s + 1, lane, T[nb][1]);
        load_frag_g(wih, (g + 16) * 8 + s + 1, lane, T[nb][2]);
      } else if (s < 11) {
        int ks = s - 7;
        load_frag_g(whh, (g)      * 4 + ks, lane, T[nb][0]);
        load_frag_g(whh, (g + 8)  * 4 + ks, lane, T[nb][1]);
        load_frag_g(whh, (g + 16) * 4 + ks, lane, T[nb][2]);
      }
      if (s < 8) {
        ir  = wmma_bf16(Aa[s].v, T[cb][0].v, ir);
        iz  = wmma_bf16(Aa[s].v, T[cb][1].v, iz);
        in_ = wmma_bf16(Aa[s].v, T[cb][2].v, in_);
      } else {
        int ks = s - 8;
        hr = wmma_bf16(Ah[ks].v, T[cb][0].v, hr);
        hz = wmma_bf16(Ah[ks].v, T[cb][1].v, hz);
        hn = wmma_bf16(Ah[ks].v, T[cb][2].v, hn);
      }
    }

    const int o = g * 16 + ncol;
    const float bir = b_ih[o], biz = b_ih[o + 128], bin = b_ih[o + 256];
    const float bhr = b_hh[o], bhz = b_hh[o + 128], bhn = b_hh[o + 256];
#pragma unroll
    for (int r = 0; r < 8; ++r) {
      long long node = nbase + mr + r;
      long long off  = node * 128 + o;
      float hval = hv_cur[off];
      float rg = sigmoid_f(ir[r] + bir + hr[r] + bhr);
      float zg = sigmoid_f(iz[r] + biz + hz[r] + bhz);
      float ng = tanhf(in_[r] + bin + rg * (hn[r] + bhn));
      float out = (1.0f - zg) * ng + zg * hval;
      hv_out[off]    = out;
      hv_bf_out[off] = f32_to_bf16_rne(out);
    }
  }
}

// ---------------- launch ----------------

extern "C" void kernel_launch(void* const* d_in, const int* in_sizes, int n_in,
                              void* d_out, int out_size, void* d_ws, size_t ws_size,
                              hipStream_t stream) {
  (void)in_sizes; (void)n_in; (void)out_size; (void)ws_size;
  const float* hv    = (const float*)d_in[0];
  const float* he    = (const float*)d_in[1];
  const int*   src   = (const int*)d_in[2];
  const int*   dst   = (const int*)d_in[3];
  const float* W_msg = (const float*)d_in[4];
  const float* b_msg = (const float*)d_in[5];
  const float* W_ih  = (const float*)d_in[6];
  const float* W_hh  = (const float*)d_in[7];
  const float* b_ih  = (const float*)d_in[8];
  const float* b_hh  = (const float*)d_in[9];

  char* ws = (char*)d_ws;
  float*          a_acc  = (float*)(ws + 0);                  // 51,200,000 B
  unsigned short* hv_bf  = (unsigned short*)(ws + 51200000);  // 12,800,000 B
  float*          hv_tmp = (float*)(ws + 64000000);           // 25,600,000 B
  unsigned short* wmsg_f = (unsigned short*)(ws + 89600000);  //    147,456 B (16nt x 9ks)
  unsigned short* wih_f  = (unsigned short*)(ws + 89747456);  //    196,608 B
  unsigned short* whh_f  = (unsigned short*)(ws + 89944064);  //     98,304 B

  k_cvt_bf16<<<(N_NODES * HDIM + 255) / 256, 256, 0, stream>>>(hv, hv_bf, N_NODES * HDIM);

  for (int t = 0; t < 2; ++t) {
    const float* Wm = W_msg + (long long)t * 256 * 257;
    const float* bm = b_msg + t * 256;
    const float* Wi = W_ih  + (long long)t * 384 * 256;
    const float* Wh = W_hh  + (long long)t * 384 * 128;
    const float* bi = b_ih  + t * 384;
    const float* bh = b_hh  + t * 384;

    k_zero4<<<(N_NODES * 256 / 4 + 255) / 256, 256, 0, stream>>>((float4*)a_acc, N_NODES * 256 / 4);
    k_pack_wmsg<<<(16 * 9 * 512 + 255) / 256, 256, 0, stream>>>(Wm, bm, wmsg_f);
    k_pack_bfrag<<<(98304 + 255) / 256, 256, 0, stream>>>(Wi, 256, 8, 98304, wih_f);
    k_pack_bfrag<<<(49152 + 255) / 256, 256, 0, stream>>>(Wh, 128, 4, 49152, whh_f);

    k_edge<<<N_EDGES / 16 / 8, 256, 0, stream>>>(hv_bf, he, src, dst, wmsg_f, a_acc);

    const float* hcur = (t == 0) ? hv : hv_tmp;
    float* hout = (t == 0) ? hv_tmp : (float*)d_out;
    k_gru<<<N_NODES / 16 / 5, 160, 0, stream>>>(a_acc, hv_bf, hcur, wih_f, whh_f, bi, bh,
                                                hout, hv_bf);
  }
}
